// Similarity_2430951489723
// MI455X (gfx1250) — matrix operand order repaired and checked by reading, not compiled, not run on hardware
//
#include <hip/hip_runtime.h>

// diag(q1 @ M @ q2^T):  N=131072 rows, D=512.
// Per-wave 16-row slab of T = q1 @ M via v_wmma_f32_16x16x32_bf16 with an
// f32 -> (bf16_hi, bf16_lo) split (3 WMMAs: hh + hl + lh) for near-f32
// accuracy, fused row-wise dot with q2, wave32 butterfly reduction.
// M tiles are cooperatively staged f32->bf16, transposed, double-buffered
// in LDS with packed ds_store_b128 writes and conflict-free b128 reads.

constexpr int Dim        = 512;
constexpr int KTILE      = 32;   // K per wmma step
constexpr int NCHUNK     = 64;   // columns per accumulation chunk (4 tiles)
constexpr int BPAD       = 40;   // padded k-stride (80B = 20 banks -> conflict-free b128)
constexpr int WAVES      = 8;
constexpr int BLOCK_ROWS = WAVES * 16;  // 128 rows per block

typedef __attribute__((ext_vector_type(16))) __bf16 v16bf;
typedef __attribute__((ext_vector_type(8)))  __bf16 v8bf;
typedef __attribute__((ext_vector_type(8)))  float  v8f;

__device__ __forceinline__ void bsplit(float x, __bf16& h, __bf16& l) {
    h = (__bf16)x;                 // RNE round to bf16
    l = (__bf16)(x - (float)h);    // residual, next 8 mantissa bits
}

__global__ __launch_bounds__(256)
void sim_bilinear_kernel(const float* __restrict__ q1,
                         const float* __restrict__ q2,
                         const float* __restrict__ M,
                         float* __restrict__ out)
{
    // Double-buffered transposed B tiles: [buf][col within chunk][k within tile]
    __shared__ __align__(16) __bf16 sBhi[2][NCHUNK][BPAD];
    __shared__ __align__(16) __bf16 sBlo[2][NCHUNK][BPAD];

    const int tid    = threadIdx.x;
    const int wave   = tid >> 5;
    const int lane   = tid & 31;
    const int nl     = lane & 15;
    const bool upper = lane >= 16;

    const int m0    = blockIdx.x * BLOCK_ROWS + wave * 16;  // wave's row slab
    const int arow  = m0 + nl;                              // A fragment row
    const int koffA = upper ? 16 : 0;   // A: K 0-15 vs 16-31
    const int koffB = upper ? 16 : 0;   // B: K 0-15 vs 16-31
    const int h8    = upper ? 8 : 0;    // C rows: v vs v+8

    // Cooperative M-tile staging map: 256 threads cover 32(k) x 64(n)
    const int sc = tid & 63;         // column within chunk
    const int sr = (tid >> 6) * 8;   // k-row base (8 rows per thread, 16B aligned)

    float acc[8];
#pragma unroll
    for (int v = 0; v < 8; ++v) acc[v] = 0.f;

    for (int nc = 0; nc < Dim / NCHUNK; ++nc) {
        const int c0 = nc * NCHUNK;

        // Stage k-tile 0 into buffer 0: coalesced f32 reads, split, one packed
        // ds_store_b128 each for hi and lo (address sc*80 + sr*2 is 16B aligned).
        {
            const float* mp = M + (size_t)sr * Dim + c0 + sc;
            v8bf hv, lv;
#pragma unroll
            for (int j = 0; j < 8; ++j) {
                __bf16 h, l;
                bsplit(mp[(size_t)j * Dim], h, l);
                hv[j] = h; lv[j] = l;
            }
            *(v8bf*)&sBhi[0][sc][sr] = hv;
            *(v8bf*)&sBlo[0][sc][sr] = lv;
        }
        __syncthreads();

        v8f C[4];
#pragma unroll
        for (int t = 0; t < 4; ++t)
#pragma unroll
            for (int i = 0; i < 8; ++i) C[t][i] = 0.f;

        for (int kk = 0; kk < Dim / KTILE; ++kk) {
            const int cur = kk & 1;
            const int k0  = kk * KTILE;

            // Stage next k-tile into the other buffer (nobody reads it now)
            if (kk + 1 < Dim / KTILE) {
                const float* mp = M + (size_t)(k0 + KTILE + sr) * Dim + c0 + sc;
                v8bf hv, lv;
#pragma unroll
                for (int j = 0; j < 8; ++j) {
                    __bf16 h, l;
                    bsplit(mp[(size_t)j * Dim], h, l);
                    hv[j] = h; lv[j] = l;
                }
                *(v8bf*)&sBhi[cur ^ 1][sc][sr] = hv;
                *(v8bf*)&sBlo[cur ^ 1][sc][sr] = lv;
            }

            // A fragment: 16 f32 of this lane's q1 row -> bf16 hi/lo
            const float4* pa = (const float4*)(q1 + (size_t)arow * Dim + k0 + koffA);
            const float4  x0 = pa[0];
            const float4  x1 = pa[1];
            const float4* pb = (const float4*)(q1 + (size_t)arow * Dim + k0 + koffA + 8);
            const float4  y0 = pb[0];
            const float4  y1 = pb[1];
            const float xs[16] = {x0.x, x0.y, x0.z, x0.w, x1.x, x1.y, x1.z, x1.w,
                                  y0.x, y0.y, y0.z, y0.w, y1.x, y1.y, y1.z, y1.w};
            v16bf ahi, alo;
#pragma unroll
            for (int i = 0; i < 16; ++i) {
                __bf16 h, l;
                bsplit(xs[i], h, l);
                ahi[i] = h;
                alo[i] = l;
            }

#pragma unroll
            for (int t = 0; t < 4; ++t) {
                const __bf16* bh = &sBhi[cur][t * 16 + nl][koffB];
                const __bf16* bl = &sBlo[cur][t * 16 + nl][koffB];
                const v8bf bh0 = *(const v8bf*)bh;
                const v8bf bh1 = *(const v8bf*)(bh + 8);
                const v8bf bl0 = *(const v8bf*)bl;
                const v8bf bl1 = *(const v8bf*)(bl + 8);
                v16bf bhi, blo;
#pragma unroll
                for (int i = 0; i < 8; ++i) {
                    bhi[i] = bh0[i]; bhi[i + 8] = bh1[i];
                    blo[i] = bl0[i]; blo[i + 8] = bl1[i];
                }
                // hi*hi + hi*lo + lo*hi  (lo*lo below f32 noise floor)
                C[t] = __builtin_amdgcn_wmma_f32_16x16x32_bf16(
                           false, ahi, false, bhi, (short)0, C[t], false, false);
                C[t] = __builtin_amdgcn_wmma_f32_16x16x32_bf16(
                           false, ahi, false, blo, (short)0, C[t], false, false);
                C[t] = __builtin_amdgcn_wmma_f32_16x16x32_bf16(
                           false, alo, false, bhi, (short)0, C[t], false, false);
            }
            __syncthreads();
        }

        // Fused dot with q2 (single-use stream: non-temporal loads).
        // C VGPR v / lane l holds T[m0 + h8 + v, c0 + t*16 + nl].
#pragma unroll
        for (int t = 0; t < 4; ++t) {
            const int col = c0 + t * 16 + nl;
#pragma unroll
            for (int v = 0; v < 8; ++v) {
                const float q2v =
                    __builtin_nontemporal_load(q2 + (size_t)(m0 + h8 + v) * Dim + col);
                acc[v] += C[t][v] * q2v;
            }
        }
    }

    // Butterfly reduce over the 16 lanes of each half-wave (masks stay in-half)
#pragma unroll
    for (int mask = 1; mask <= 8; mask <<= 1) {
#pragma unroll
        for (int v = 0; v < 8; ++v)
            acc[v] += __shfl_xor(acc[v], mask, 32);
    }
    if (nl == 0) {  // lanes 0 and 16 write their 8 rows each
#pragma unroll
        for (int v = 0; v < 8; ++v)
            __builtin_nontemporal_store(acc[v], out + m0 + h8 + v);
    }
}

extern "C" void kernel_launch(void* const* d_in, const int* in_sizes, int n_in,
                              void* d_out, int out_size, void* d_ws, size_t ws_size,
                              hipStream_t stream) {
    const float* q1  = (const float*)d_in[0];
    const float* q2  = (const float*)d_in[1];
    const float* M   = (const float*)d_in[2];
    float*       out = (float*)d_out;

    const int N    = in_sizes[0] / Dim;        // 131072
    const int grid = N / BLOCK_ROWS;           // 1024 blocks of 256 threads

    hipLaunchKernelGGL(sim_bilinear_kernel, dim3(grid), dim3(256), 0, stream,
                       q1, q2, M, out);
}
